// VAE_28999619183191
// MI455X (gfx1250) — matrix operand order repaired and checked by reading, not compiled, not run on hardware
//
#include <hip/hip_runtime.h>
#include <hip/hip_bf16.h>
#include <math.h>

#define N_NODES 50000
#define N_EDGES 800000
#define IN_F 64
#define H_F 128
#define OUT_F 64
#define NS_S 10
#define KSEL 25000
#define SORT_M 65536

typedef __attribute__((ext_vector_type(16))) _Float16 v16h;
typedef __attribute__((ext_vector_type(8)))  _Float16 v8h;
typedef __attribute__((ext_vector_type(8)))  float    v8f;

// ---------------------------------------------------------------------------
// WMMA fragment helpers (CDNA5 v_wmma_f32_16x16x32_f16 layouts)
__device__ __forceinline__ v16h load_a_frag_h(const _Float16* __restrict__ A,
                                              int lda, int mBase, int kBase) {
  const int lane = threadIdx.x & 31;
  const int half = lane >> 4;
  const int m    = lane & 15;
  const _Float16* p = A + (size_t)(mBase + m) * lda + kBase + half * 8;
  v8h lo = *(const v8h*)p;         // K = kBase + half*8 .. +7
  v8h hi = *(const v8h*)(p + 16);  // K = kBase + 16 + half*8 .. +7
  return __builtin_shufflevector(lo, hi, 0, 1, 2, 3, 4, 5, 6, 7,
                                 8, 9, 10, 11, 12, 13, 14, 15);
}

__device__ __forceinline__ v16h load_b_packed(const _Float16* __restrict__ Wp,
                                              int kTiles, int nTile, int kt) {
  const int lane = threadIdx.x & 31;
  const _Float16* p = Wp + (((size_t)nTile * kTiles + kt) * 32 + lane) * 16;
  v8h lo = *(const v8h*)p;
  v8h hi = *(const v8h*)(p + 8);
  return __builtin_shufflevector(lo, hi, 0, 1, 2, 3, 4, 5, 6, 7,
                                 8, 9, 10, 11, 12, 13, 14, 15);
}

// Repack f32 weight (K x Nout row-major) into fragment-ordered f16.
__global__ void repack_w(const float* __restrict__ W, _Float16* __restrict__ Wp,
                         int K, int Nout) {
  int t = blockIdx.x * blockDim.x + threadIdx.x;
  int kTiles = K >> 5, nTiles = Nout >> 4;
  if (t >= nTiles * kTiles * 32) return;
  int lane = t & 31, tile = t >> 5;
  int kt = tile % kTiles, nt = tile / kTiles;
  int half = lane >> 4, n = lane & 15;
  int kBase = kt * 32, nBase = nt * 16;
  _Float16* p = Wp + (size_t)t * 16;
#pragma unroll
  for (int r = 0; r < 8; ++r) {
    int k = half * 8 + 2 * r + (r >= 4 ? 8 : 0);
    p[2 * r]     = (_Float16)W[(size_t)(kBase + k) * Nout + nBase + n];
    p[2 * r + 1] = (_Float16)W[(size_t)(kBase + k + 1) * Nout + nBase + n];
  }
}

// f32 -> f16 activation conversion, optional per-row scale (1/deg fused).
__global__ void conv_h(const float* __restrict__ in, const float* __restrict__ rs,
                       _Float16* __restrict__ out, long n, int C) {
  long t = (long)blockIdx.x * blockDim.x + threadIdx.x;
  if (t >= n) return;
  float s = rs ? rs[t / C] : 1.0f;
  out[t] = (_Float16)(in[t] * s);
}

// C = relu?( A1 @ W1 [+ A2 @ W2] + bias ), f16 in, f32 accumulate.
// KT = K/32 (compile-time -> fully unrolled, software-pipelined fragments).
// Wave computes 16M x 64N (4 n-tiles); blockDim=128 (4 waves = 4 m-tiles).
template <int KT, bool DUAL>
__global__ void gemm_h(const _Float16* __restrict__ A1, const _Float16* __restrict__ W1p,
                       const _Float16* __restrict__ A2, const _Float16* __restrict__ W2p,
                       const float* __restrict__ bias, float* __restrict__ C,
                       _Float16* __restrict__ Ch, int Nout, int Mtiles, int relu) {
  const int wave  = threadIdx.x >> 5;
  const int lane  = threadIdx.x & 31;
  const int half  = lane >> 4;
  const int mTile = blockIdx.x * 4 + wave;
  if (mTile >= Mtiles) return;  // wave-uniform: EXEC stays all-1 for WMMA
  const int mBase  = mTile * 16;
  const int nTile0 = blockIdx.y * 4;
  constexpr int K = KT * 32;

  v8f acc[4] = {{}, {}, {}, {}};
  // current fragment set (prologue loads)
  v16h a1c, a2c, b1c[4], b2c[4];
  a1c = load_a_frag_h(A1, K, mBase, 0);
#pragma unroll
  for (int j = 0; j < 4; ++j) b1c[j] = load_b_packed(W1p, KT, nTile0 + j, 0);
  if (DUAL) {
    a2c = load_a_frag_h(A2, K, mBase, 0);
#pragma unroll
    for (int j = 0; j < 4; ++j) b2c[j] = load_b_packed(W2p, KT, nTile0 + j, 0);
  }

#pragma unroll
  for (int kt = 0; kt < KT; ++kt) {
    // issue next-step loads BEFORE this step's WMMA chain (latency hiding)
    v16h a1n, a2n, b1n[4], b2n[4];
    if (kt + 1 < KT) {
      a1n = load_a_frag_h(A1, K, mBase, (kt + 1) * 32);
#pragma unroll
      for (int j = 0; j < 4; ++j) b1n[j] = load_b_packed(W1p, KT, nTile0 + j, kt + 1);
      if (DUAL) {
        a2n = load_a_frag_h(A2, K, mBase, (kt + 1) * 32);
#pragma unroll
        for (int j = 0; j < 4; ++j) b2n[j] = load_b_packed(W2p, KT, nTile0 + j, kt + 1);
      }
    }
#pragma unroll
    for (int j = 0; j < 4; ++j)
      acc[j] = __builtin_amdgcn_wmma_f32_16x16x32_f16(false, a1c, false, b1c[j],
                                                      (short)0, acc[j], false, false);
    if (DUAL) {
#pragma unroll
      for (int j = 0; j < 4; ++j)
        acc[j] = __builtin_amdgcn_wmma_f32_16x16x32_f16(false, a2c, false, b2c[j],
                                                        (short)0, acc[j], false, false);
    }
    if (kt + 1 < KT) {
      a1c = a1n;
#pragma unroll
      for (int j = 0; j < 4; ++j) b1c[j] = b1n[j];
      if (DUAL) {
        a2c = a2n;
#pragma unroll
        for (int j = 0; j < 4; ++j) b2c[j] = b2n[j];
      }
    }
  }

  // C/D layout: VGPR v -> M = v + 8*half, N = lane&15
  const int n = lane & 15;
#pragma unroll
  for (int j = 0; j < 4; ++j) {
    const int nB = (nTile0 + j) * 16;
    const float bv = bias ? bias[nB + n] : 0.0f;
#pragma unroll
    for (int v = 0; v < 8; ++v) {
      float val = acc[j][v] + bv;
      if (relu) val = fmaxf(val, 0.0f);
      size_t idx = (size_t)(mBase + v + 8 * half) * Nout + nB + n;
      if (C)  C[idx]  = val;
      if (Ch) Ch[idx] = (_Float16)val;
    }
  }
}

// ---------------- graph / elementwise kernels ------------------------------
__global__ void zero_f(float* p, long n) {
  long t = (long)blockIdx.x * blockDim.x + threadIdx.x;
  if (t < n) p[t] = 0.0f;
}

__global__ void degree_k(const int* __restrict__ dst, float* deg, int E) {
  int e = blockIdx.x * blockDim.x + threadIdx.x;
  if (e < E) atomicAdd(&deg[dst[e]], 1.0f);
}

__global__ void invdeg_k(const float* deg, float* inv, int n) {
  int i = blockIdx.x * blockDim.x + threadIdx.x;
  if (i < n) inv[i] = 1.0f / fmaxf(deg[i], 1.0f);
}

// float4-vectorized scatter: one (src,dst) index load per 4 atomics.
__global__ void scatter_add4(const float* __restrict__ x, const int* __restrict__ src,
                             const int* __restrict__ dst, float* agg, int E, int F4) {
  long t = (long)blockIdx.x * blockDim.x + threadIdx.x;
  if (t >= (long)E * F4) return;
  int e = (int)(t / F4), c4 = (int)(t % F4);
  const float4 v = ((const float4*)(x + (size_t)src[e] * F4 * 4))[c4];
  float* a = agg + (size_t)dst[e] * F4 * 4 + (size_t)c4 * 4;
  atomicAdd(a + 0, v.x); atomicAdd(a + 1, v.y);
  atomicAdd(a + 2, v.z); atomicAdd(a + 3, v.w);
}

__global__ void bn_stats(const float* __restrict__ x, float* sum, float* sq,
                         int Nrows, int C, int chunk) {
  int c = blockIdx.y * blockDim.x + threadIdx.x;
  if (c >= C) return;
  long r0 = (long)blockIdx.x * chunk;
  long r1 = r0 + chunk; if (r1 > Nrows) r1 = Nrows;
  float s = 0.f, q = 0.f;
  for (long r = r0; r < r1; ++r) { float v = x[r * C + c]; s += v; q += v * v; }
  atomicAdd(&sum[c], s); atomicAdd(&sq[c], q);
}

__global__ void bn_apply(const float* __restrict__ in, float* __restrict__ out,
                         const float* sum, const float* sq,
                         const float* gamma, const float* beta, long Nrows, int C) {
  long t = (long)blockIdx.x * blockDim.x + threadIdx.x;
  if (t >= Nrows * C) return;
  int c = (int)(t % C);
  float m  = sum[c] / (float)Nrows;
  float va = sq[c] / (float)Nrows - m * m;
  float sc = gamma[c] * rsqrtf(va + 1e-5f);
  out[t] = (in[t] - m) * sc + beta[c];
}

__global__ void score_k(const float* __restrict__ agg, const float* __restrict__ x1,
                        const float* __restrict__ Wrel, const float* __restrict__ bp,
                        const float* __restrict__ Wroot, float* score, int n) {
  int i = blockIdx.x * blockDim.x + threadIdx.x;
  if (i >= n) return;
  float s = bp[0];
  const float* ar = agg + (size_t)i * H_F;
  const float* xr = x1  + (size_t)i * H_F;
#pragma unroll 4
  for (int c = 0; c < H_F; ++c) s += ar[c] * Wrel[c] + xr[c] * Wroot[c];
  score[i] = tanhf(s);
}

__global__ void z_k(const float* __restrict__ ms, const float* __restrict__ noise,
                    float* mean_o, float* ls_o, _Float16* zh, int n) {
  long t = (long)blockIdx.x * blockDim.x + threadIdx.x;
  if (t >= (long)n * OUT_F) return;
  int i = (int)(t >> 6), c = (int)(t & 63);
  float m  = ms[(size_t)i * 128 + c];
  float ls = ms[(size_t)i * 128 + 64 + c];
  float nm = 0.f;
#pragma unroll
  for (int s = 0; s < NS_S; ++s) nm += noise[(size_t)s * n * OUT_F + t];
  nm *= (1.0f / NS_S);
  mean_o[t] = m; ls_o[t] = ls;
  zh[t] = (_Float16)(m + expf(ls) * nm);
}

// ---------------- top-k via global bitonic sort (descending) ---------------
__global__ void sort_init(const float* score, float* key, int* idx) {
  int i = blockIdx.x * blockDim.x + threadIdx.x;
  if (i >= SORT_M) return;
  key[i] = (i < N_NODES) ? score[i] : -3.4e38f;
  idx[i] = i;
}

__global__ void sort_step(float* key, int* idx, int j, int kk) {
  int i = blockIdx.x * blockDim.x + threadIdx.x;
  if (i >= SORT_M) return;
  int l = i ^ j;
  if (l > i) {
    float ki = key[i], kl = key[l];
    bool desc = ((i & kk) == 0);
    bool sw = desc ? (ki < kl) : (ki > kl);
    if (sw) { key[i] = kl; key[l] = ki; int t = idx[i]; idx[i] = idx[l]; idx[l] = t; }
  }
}

__global__ void mark_keep(const int* sidx, int* keep, int k) {
  int r = blockIdx.x * blockDim.x + threadIdx.x;
  if (r < k) keep[sidx[r]] = 1;
}

// perm[newidx[v]] == v for kept v => unpooled edge = original index if kept
__global__ void edge_out_k(const int* __restrict__ src, const int* __restrict__ dst,
                           const int* __restrict__ keep, int* ei_out, int* mask_out, int E) {
  int e = blockIdx.x * blockDim.x + threadIdx.x;
  if (e >= E) return;
  int s = src[e], d = dst[e];
  int m = keep[s] & keep[d];
  mask_out[e]   = m;
  ei_out[e]     = m ? s : -1;
  ei_out[E + e] = m ? d : -1;
}

__global__ void gather_rows(const float* __restrict__ x1, const int* __restrict__ sidx,
                            float* outp, int k, int C) {
  long t = (long)blockIdx.x * blockDim.x + threadIdx.x;
  if (t >= (long)k * C) return;
  int r = (int)(t / C), c = (int)(t % C);
  outp[t] = x1[(size_t)sidx[r] * C + c];
}

// ---------------------------------------------------------------------------
extern "C" void kernel_launch(void* const* d_in, const int* in_sizes, int n_in,
                              void* d_out, int out_size, void* d_ws, size_t ws_size,
                              hipStream_t stream) {
  const float* x      = (const float*)d_in[0];
  const int*   ei     = (const int*)  d_in[1];
  const float* noise  = (const float*)d_in[3];
  const float* We1l = (const float*)d_in[4];  const float* be1 = (const float*)d_in[5];
  const float* We1r = (const float*)d_in[6];  const float* g1  = (const float*)d_in[7];
  const float* b1   = (const float*)d_in[8];
  const float* We2l = (const float*)d_in[9];  const float* be2 = (const float*)d_in[10];
  const float* We2r = (const float*)d_in[11]; const float* g2  = (const float*)d_in[12];
  const float* b2   = (const float*)d_in[13];
  const float* We3l = (const float*)d_in[14]; const float* be3 = (const float*)d_in[15];
  const float* We3r = (const float*)d_in[16];
  const float* Wt   = (const float*)d_in[17]; const float* bt  = (const float*)d_in[18];
  const float* Wd1  = (const float*)d_in[19]; const float* bd1 = (const float*)d_in[20];
  const float* gd   = (const float*)d_in[21]; const float* bd  = (const float*)d_in[22];
  const float* Wd2  = (const float*)d_in[23]; const float* bd2 = (const float*)d_in[24];
  const float* Wprel = (const float*)d_in[25]; const float* bp  = (const float*)d_in[26];
  const float* Wproot = (const float*)d_in[27];

  const int* src = ei;
  const int* dst = ei + N_EDGES;

  // d_out layout (flat, return order)
  size_t off = 0;
  float* o_recon = (float*)d_out;                      off += (size_t)N_NODES * IN_F;
  float* o_xun   = (float*)d_out + off;                off += (size_t)KSEL * H_F;
  int*   o_ei    = (int*)((float*)d_out + off);        off += (size_t)2 * N_EDGES;
  int*   o_mask  = (int*)((float*)d_out + off);        off += (size_t)N_EDGES;
  float* o_mean  = (float*)d_out + off;                off += (size_t)N_NODES * OUT_F;
  float* o_ls    = (float*)d_out + off;                off += (size_t)N_NODES * OUT_F;
  float* o_x1    = (float*)d_out + off;

  // workspace bump allocator
  char* w = (char*)d_ws;
  auto alloc = [&](size_t bytes) -> void* {
    void* p = (void*)w; w += (bytes + 255) & ~(size_t)255; return p;
  };
  // f32 buffers
  float* deg   = (float*)alloc((size_t)N_NODES * 4);
  float* invd  = (float*)alloc((size_t)N_NODES * 4);
  float* agg1  = (float*)alloc((size_t)N_NODES * IN_F * 4);
  float* tmp1  = (float*)alloc((size_t)N_NODES * H_F * 4);     // x1_pre
  float* agg2  = (float*)alloc((size_t)N_NODES * H_F * 4);     // raw; reused as h(f32)
  float* x2    = (float*)alloc((size_t)N_NODES * 2 * H_F * 4);
  float* agg3  = (float*)alloc((size_t)N_NODES * 2 * H_F * 4);
  float* msbuf = (float*)alloc((size_t)N_NODES * 2 * OUT_F * 4);
  float* score = (float*)alloc((size_t)N_NODES * 4);
  float* skey  = (float*)alloc((size_t)SORT_M * 4);
  int*   sidx  = (int*)  alloc((size_t)SORT_M * 4);
  int*   keep  = (int*)  alloc((size_t)N_NODES * 4);
  float* bnsum = (float*)alloc(256 * 4);
  float* bnsq  = (float*)alloc(256 * 4);
  // f16 activation buffers (with lifetime-based aliases)
  _Float16* x_h    = (_Float16*)alloc((size_t)N_NODES * IN_F * 2);
  _Float16* agg1_h = (_Float16*)alloc((size_t)N_NODES * IN_F * 2);
  _Float16* x1_h   = (_Float16*)alloc((size_t)N_NODES * H_F * 2);
  _Float16* agg2_h = (_Float16*)alloc((size_t)N_NODES * H_F * 2);
  _Float16* x2_h   = (_Float16*)alloc((size_t)N_NODES * 2 * H_F * 2);
  _Float16* agg3_h = (_Float16*)alloc((size_t)N_NODES * 2 * H_F * 2);
  _Float16* z_h   = x_h;     // free after layer 1
  _Float16* x1t_h = agg1_h;  // free after layer 1
  _Float16* h_h   = agg2_h;  // free after layer 2
  float*    hbuf  = agg2;    // raw agg2 f32 free after score
  // fragment-ordered f16 weights
  _Float16* wp_e1l = (_Float16*)alloc((size_t)IN_F * H_F * 2);
  _Float16* wp_e1r = (_Float16*)alloc((size_t)IN_F * H_F * 2);
  _Float16* wp_e2l = (_Float16*)alloc((size_t)H_F * 2 * H_F * 2);
  _Float16* wp_e2r = (_Float16*)alloc((size_t)H_F * 2 * H_F * 2);
  _Float16* wp_e3l = (_Float16*)alloc((size_t)2 * H_F * 2 * OUT_F * 2);
  _Float16* wp_e3r = (_Float16*)alloc((size_t)2 * H_F * 2 * OUT_F * 2);
  _Float16* wp_t   = (_Float16*)alloc((size_t)H_F * IN_F * 2);
  _Float16* wp_d1a = (_Float16*)alloc((size_t)OUT_F * H_F * 2);
  _Float16* wp_d1b = (_Float16*)alloc((size_t)IN_F * H_F * 2);
  _Float16* wp_d2  = (_Float16*)alloc((size_t)H_F * IN_F * 2);

  const int MT = N_NODES / 16;            // 3125 exact
  const dim3 gB((MT + 3) / 4);            // 4 m-tile waves per block
  auto eb = [](long n) { return dim3((unsigned)((n + 255) / 256)); };

  // weight repacks (tiny; L2-resident afterwards)
  repack_w<<<eb(8 * 2 * 32), 256, 0, stream>>>(We1l, wp_e1l, IN_F, H_F);
  repack_w<<<eb(8 * 2 * 32), 256, 0, stream>>>(We1r, wp_e1r, IN_F, H_F);
  repack_w<<<eb(16 * 4 * 32), 256, 0, stream>>>(We2l, wp_e2l, H_F, 2 * H_F);
  repack_w<<<eb(16 * 4 * 32), 256, 0, stream>>>(We2r, wp_e2r, H_F, 2 * H_F);
  repack_w<<<eb(8 * 8 * 32), 256, 0, stream>>>(We3l, wp_e3l, 2 * H_F, 2 * OUT_F);
  repack_w<<<eb(8 * 8 * 32), 256, 0, stream>>>(We3r, wp_e3r, 2 * H_F, 2 * OUT_F);
  repack_w<<<eb(4 * 4 * 32), 256, 0, stream>>>(Wt, wp_t, H_F, IN_F);
  repack_w<<<eb(8 * 2 * 32), 256, 0, stream>>>(Wd1, wp_d1a, OUT_F, H_F);
  repack_w<<<eb(8 * 2 * 32), 256, 0, stream>>>(Wd1 + (size_t)OUT_F * H_F, wp_d1b, IN_F, H_F);
  repack_w<<<eb(4 * 4 * 32), 256, 0, stream>>>(Wd2, wp_d2, H_F, IN_F);

  // degree + inverse degree
  zero_f<<<eb(N_NODES), 256, 0, stream>>>(deg, N_NODES);
  degree_k<<<eb(N_EDGES), 256, 0, stream>>>(dst, deg, N_EDGES);
  invdeg_k<<<eb(N_NODES), 256, 0, stream>>>(deg, invd, N_NODES);

  // ---- SAGE layer 1
  zero_f<<<eb((long)N_NODES * IN_F), 256, 0, stream>>>(agg1, (long)N_NODES * IN_F);
  scatter_add4<<<eb((long)N_EDGES * (IN_F / 4)), 256, 0, stream>>>(x, src, dst, agg1, N_EDGES, IN_F / 4);
  conv_h<<<eb((long)N_NODES * IN_F), 256, 0, stream>>>(agg1, invd, agg1_h, (long)N_NODES * IN_F, IN_F);
  conv_h<<<eb((long)N_NODES * IN_F), 256, 0, stream>>>(x, nullptr, x_h, (long)N_NODES * IN_F, IN_F);
  gemm_h<2, true><<<dim3(gB.x, 2), 128, 0, stream>>>(agg1_h, wp_e1l, x_h, wp_e1r, be1,
                                                     tmp1, nullptr, H_F, MT, 1);
  zero_f<<<dim3(2), 256, 0, stream>>>(bnsum, 256);
  zero_f<<<dim3(2), 256, 0, stream>>>(bnsq, 256);
  bn_stats<<<dim3(98, 1), 128, 0, stream>>>(tmp1, bnsum, bnsq, N_NODES, H_F, 512);
  bn_apply<<<eb((long)N_NODES * H_F), 256, 0, stream>>>(tmp1, o_x1, bnsum, bnsq, g1, b1, N_NODES, H_F);
  conv_h<<<eb((long)N_NODES * H_F), 256, 0, stream>>>(o_x1, nullptr, x1_h, (long)N_NODES * H_F, H_F);

  // ---- SAGE layer 2 (agg2 kept raw in f32 for pooling score)
  zero_f<<<eb((long)N_NODES * H_F), 256, 0, stream>>>(agg2, (long)N_NODES * H_F);
  scatter_add4<<<eb((long)N_EDGES * (H_F / 4)), 256, 0, stream>>>(o_x1, src, dst, agg2, N_EDGES, H_F / 4);
  conv_h<<<eb((long)N_NODES * H_F), 256, 0, stream>>>(agg2, invd, agg2_h, (long)N_NODES * H_F, H_F);
  gemm_h<4, true><<<dim3(gB.x, 4), 128, 0, stream>>>(agg2_h, wp_e2l, x1_h, wp_e2r, be2,
                                                     x2, nullptr, 2 * H_F, MT, 1);
  zero_f<<<dim3(2), 256, 0, stream>>>(bnsum, 256);
  zero_f<<<dim3(2), 256, 0, stream>>>(bnsq, 256);
  bn_stats<<<dim3(98, 2), 128, 0, stream>>>(x2, bnsum, bnsq, N_NODES, 2 * H_F, 512);
  bn_apply<<<eb((long)N_NODES * 2 * H_F), 256, 0, stream>>>(x2, x2, bnsum, bnsq, g2, b2, N_NODES, 2 * H_F);
  conv_h<<<eb((long)N_NODES * 2 * H_F), 256, 0, stream>>>(x2, nullptr, x2_h, (long)N_NODES * 2 * H_F, 2 * H_F);

  // pooling score (raw agg2 + x1)
  score_k<<<eb(N_NODES), 256, 0, stream>>>(agg2, o_x1, Wprel, bp, Wproot, score, N_NODES);

  // ---- SAGE layer 3 -> ms (mean | log_std)
  zero_f<<<eb((long)N_NODES * 2 * H_F), 256, 0, stream>>>(agg3, (long)N_NODES * 2 * H_F);
  scatter_add4<<<eb((long)N_EDGES * (2 * H_F / 4)), 256, 0, stream>>>(x2, src, dst, agg3, N_EDGES, 2 * H_F / 4);
  conv_h<<<eb((long)N_NODES * 2 * H_F), 256, 0, stream>>>(agg3, invd, agg3_h, (long)N_NODES * 2 * H_F, 2 * H_F);
  gemm_h<8, true><<<dim3(gB.x, 2), 128, 0, stream>>>(agg3_h, wp_e3l, x2_h, wp_e3r, be3,
                                                     msbuf, nullptr, 2 * OUT_F, MT, 0);

  // reparameterize (writes mean/log_std f32 + z f16)
  z_k<<<eb((long)N_NODES * OUT_F), 256, 0, stream>>>(msbuf, noise, o_mean, o_ls, z_h, N_NODES);

  // decoder: x1t = x1@Wt+bt ; h = BN(relu([z,x1t]@Wd1+bd1)) ; recon = h@Wd2+bd2
  gemm_h<4, false><<<dim3(gB.x, 1), 128, 0, stream>>>(x1_h, wp_t, nullptr, nullptr, bt,
                                                      nullptr, x1t_h, IN_F, MT, 0);
  gemm_h<2, true><<<dim3(gB.x, 2), 128, 0, stream>>>(z_h, wp_d1a, x1t_h, wp_d1b, bd1,
                                                     hbuf, nullptr, H_F, MT, 1);
  zero_f<<<dim3(2), 256, 0, stream>>>(bnsum, 256);
  zero_f<<<dim3(2), 256, 0, stream>>>(bnsq, 256);
  bn_stats<<<dim3(98, 1), 128, 0, stream>>>(hbuf, bnsum, bnsq, N_NODES, H_F, 512);
  bn_apply<<<eb((long)N_NODES * H_F), 256, 0, stream>>>(hbuf, hbuf, bnsum, bnsq, gd, bd, N_NODES, H_F);
  conv_h<<<eb((long)N_NODES * H_F), 256, 0, stream>>>(hbuf, nullptr, h_h, (long)N_NODES * H_F, H_F);
  gemm_h<4, false><<<dim3(gB.x, 1), 128, 0, stream>>>(h_h, wp_d2, nullptr, nullptr, bd2,
                                                      o_recon, nullptr, IN_F, MT, 0);

  // ---- top-k: bitonic sort (descending) of 65536 padded (score, idx) pairs
  sort_init<<<eb(SORT_M), 256, 0, stream>>>(score, skey, sidx);
  for (int kk = 2; kk <= SORT_M; kk <<= 1)
    for (int j = kk >> 1; j > 0; j >>= 1)
      sort_step<<<eb(SORT_M), 256, 0, stream>>>(skey, sidx, j, kk);

  zero_f<<<eb(N_NODES), 256, 0, stream>>>((float*)keep, N_NODES);
  mark_keep<<<eb(KSEL), 256, 0, stream>>>(sidx, keep, KSEL);
  edge_out_k<<<eb(N_EDGES), 256, 0, stream>>>(src, dst, keep, o_ei, o_mask, N_EDGES);
  gather_rows<<<eb((long)KSEL * H_F), 256, 0, stream>>>(o_x1, sidx, o_xun, KSEL, H_F);
}
